// CuteInferMLP_82523501626096
// MI455X (gfx1250) — compile-verified, matrix operand order checked by reading, not run on hardware
//
#include <hip/hip_runtime.h>
#include <hip/hip_bf16.h>

// ---------------------------------------------------------------------------
// CuteInferMLP forward for gfx1250 (MI455X):
//   D = gelu(X @ W0^T + b0)   (fp16, in d_ws)
//   E = D @ W1^T + b1         (fp16-rounded, to d_out)
// Fast path: one-time fp32->fp16 conversion of X/W0/W1 into d_ws, then both
// GEMMs run fully on fp16 operands fetched with global_load_async_to_lds_b128
// (ASYNCcnt) straight into double-buffered LDS — zero inner-loop cvt VALU.
// Compute: v_wmma_f32_16x16x32_f16, 8 wave32s per block, 128x128x32 tiles.
// ---------------------------------------------------------------------------

typedef _Float16 v16h __attribute__((ext_vector_type(16)));
typedef _Float16 v8h  __attribute__((ext_vector_type(8)));
typedef _Float16 v4h  __attribute__((ext_vector_type(4)));
typedef float    v8f  __attribute__((ext_vector_type(8)));

#define BM 128
#define BN 128
#define BK 32
#define NTHREADS 256   // 8 wave32s: 2 (M) x 4 (N)

union V16U { v16h v; v8h h[2]; };

// one-time elementwise fp32 -> fp16 (vectorized x4)
__global__ __launch_bounds__(256)
void cvt_f32_f16(const float* __restrict__ src, _Float16* __restrict__ dst, int n4)
{
    int i = blockIdx.x * blockDim.x + threadIdx.x;
    if (i < n4) {
        float4 f = ((const float4*)src)[i];
        v4h h = { (_Float16)f.x, (_Float16)f.y, (_Float16)f.z, (_Float16)f.w };
        ((v4h*)dst)[i] = h;
    }
}

// NT GEMM: C[m][n] = sum_k A[m][k] * B[n][k]
// *_HALF: operand is fp16 -> async copy global->LDS. Else fp32 -> staged + cvt.
template<bool A_HALF, bool B_HALF, bool DO_GELU>
__global__ __launch_bounds__(NTHREADS)
void mlp_gemm_nt(const void* __restrict__ Aany,
                 const void* __restrict__ Bany,
                 const float* __restrict__ bias,
                 void* __restrict__ Outany,
                 int Kdim, int Ndim)
{
    __shared__ __align__(16) _Float16 As[2][BM * BK];
    __shared__ __align__(16) _Float16 Bs[2][BN * BK];

    const int tid   = threadIdx.x;
    const int lane  = tid & 31;
    const int wave  = tid >> 5;
    const int waveM = wave >> 2;   // 0..1
    const int waveN = wave & 3;    // 0..3

    const int m0 = blockIdx.y * BM;
    const int n0 = blockIdx.x * BN;

    const float*    Afp = (const float*)Aany;
    const _Float16* Ahp = (const _Float16*)Aany;
    const float*    Bfp = (const float*)Bany;
    const _Float16* Bhp = (const _Float16*)Bany;

    float4 stA[4], stB[4];     // staging (fp32 paths only; dead otherwise)

    // ---- stage-(kt) issue ---------------------------------------------------
    auto issueA = [&](int kt, int buf) {
        const int k0 = kt * BK;
        if (A_HALF) {
            #pragma unroll
            for (int i = 0; i < 2; ++i) {          // 128 rows x 4 chunks of 8 halves
                int idx = tid + i * NTHREADS;
                int r = idx >> 2, c = (idx & 3) * 8;
                unsigned ldsOff = (unsigned)(uintptr_t)&As[buf][r * BK + c];
                const _Float16* gp = Ahp + (size_t)(m0 + r) * Kdim + k0 + c;
                asm volatile("global_load_async_to_lds_b128 %0, %1, off"
                             :: "v"(ldsOff), "v"(gp) : "memory");
            }
        } else {
            #pragma unroll
            for (int i = 0; i < 4; ++i) {          // 128 rows x 8 float4 chunks
                int idx = tid + i * NTHREADS;
                int r = idx >> 3, c = (idx & 7) * 4;
                stA[i] = *(const float4*)(Afp + (size_t)(m0 + r) * Kdim + k0 + c);
            }
        }
    };
    auto issueB = [&](int kt, int buf) {
        const int k0 = kt * BK;
        if (B_HALF) {
            #pragma unroll
            for (int i = 0; i < 2; ++i) {
                int idx = tid + i * NTHREADS;
                int r = idx >> 2, c = (idx & 3) * 8;
                unsigned ldsOff = (unsigned)(uintptr_t)&Bs[buf][r * BK + c];
                const _Float16* gp = Bhp + (size_t)(n0 + r) * Kdim + k0 + c;
                asm volatile("global_load_async_to_lds_b128 %0, %1, off"
                             :: "v"(ldsOff), "v"(gp) : "memory");
            }
        } else {
            #pragma unroll
            for (int i = 0; i < 4; ++i) {
                int idx = tid + i * NTHREADS;
                int r = idx >> 3, c = (idx & 7) * 4;
                stB[i] = *(const float4*)(Bfp + (size_t)(n0 + r) * Kdim + k0 + c);
            }
        }
    };

    // ---- wait + cvt + LDS store (fp32 paths only) --------------------------
    auto storeA = [&](int buf) {
        if (!A_HALF) {
            #pragma unroll
            for (int i = 0; i < 4; ++i) {
                int idx = tid + i * NTHREADS;
                int r = idx >> 3, c = (idx & 7) * 4;
                v4h h = { (_Float16)stA[i].x, (_Float16)stA[i].y,
                          (_Float16)stA[i].z, (_Float16)stA[i].w };
                *(v4h*)&As[buf][r * BK + c] = h;
            }
        }
    };
    auto storeB = [&](int buf) {
        if (!B_HALF) {
            #pragma unroll
            for (int i = 0; i < 4; ++i) {
                int idx = tid + i * NTHREADS;
                int r = idx >> 3, c = (idx & 7) * 4;
                v4h h = { (_Float16)stB[i].x, (_Float16)stB[i].y,
                          (_Float16)stB[i].z, (_Float16)stB[i].w };
                *(v4h*)&Bs[buf][r * BK + c] = h;
            }
        }
    };

    v8f acc[4][2];
    #pragma unroll
    for (int mi = 0; mi < 4; ++mi)
        #pragma unroll
        for (int ni = 0; ni < 2; ++ni)
            acc[mi][ni] = (v8f)(0.0f);

    // CDNA5 WMMA fragment addressing (16-bit A 16x32 / B 32x16):
    // lane[0..15] -> row/col = lane,    k-halves {0..7, 16..23}
    // lane[16..31]-> row/col = lane-16, k-halves {8..15, 24..31}
    const int fr = lane & 15;
    const int kb = (lane >> 4) * 8;

    auto compute = [&](int cur) {
        V16U a[4], b[2];
        #pragma unroll
        for (int mi = 0; mi < 4; ++mi) {
            const _Float16* p = &As[cur][(waveM * 64 + mi * 16 + fr) * BK + kb];
            a[mi].h[0] = *(const v8h*)p;
            a[mi].h[1] = *(const v8h*)(p + 16);
        }
        #pragma unroll
        for (int ni = 0; ni < 2; ++ni) {
            const _Float16* p = &Bs[cur][(waveN * 32 + ni * 16 + fr) * BK + kb];
            b[ni].h[0] = *(const v8h*)p;
            b[ni].h[1] = *(const v8h*)(p + 16);
        }
        #pragma unroll
        for (int mi = 0; mi < 4; ++mi)
            #pragma unroll
            for (int ni = 0; ni < 2; ++ni)
                acc[mi][ni] = __builtin_amdgcn_wmma_f32_16x16x32_f16(
                    false, a[mi].v, false, b[ni].v,
                    (short)0, acc[mi][ni], false, false);
    };

    // ---- pipelined main loop -----------------------------------------------
    const int KT = Kdim / BK;
    issueA(0, 0); issueB(0, 0);
    storeA(0); storeB(0);
    if (A_HALF || B_HALF) asm volatile("s_wait_asynccnt 0x0" ::: "memory");
    __syncthreads();

    #pragma unroll 1
    for (int kt = 0; kt < KT; ++kt) {
        const int cur = kt & 1;
        const bool more = (kt + 1 < KT);
        if (more) { issueA(kt + 1, cur ^ 1); issueB(kt + 1, cur ^ 1); }
        compute(cur);                       // WMMAs overlap in-flight copies
        if (more) {
            storeA(cur ^ 1); storeB(cur ^ 1);
            if (A_HALF || B_HALF) asm volatile("s_wait_asynccnt 0x0" ::: "memory");
        }
        __syncthreads();
    }

    // ---- epilogue -----------------------------------------------------------
    // C/D layout: VGPR v, lane<16 -> (m=v, n=lane); lane>=16 -> (m=v+8, n=lane-16)
    const int mhi = (lane >> 4) * 8;
    const int nn  = lane & 15;
    #pragma unroll
    for (int mi = 0; mi < 4; ++mi) {
        #pragma unroll
        for (int ni = 0; ni < 2; ++ni) {
            const int n  = n0 + waveN * 32 + ni * 16 + nn;
            const float bv = bias[n];
            #pragma unroll
            for (int v = 0; v < 8; ++v) {
                const int m = m0 + waveM * 64 + mi * 16 + v + mhi;
                float x = acc[mi][ni][v] + bv;
                if (DO_GELU) {
                    x = 0.5f * x * (1.0f + erff(x * 0.70710678118654752f));
                    ((_Float16*)Outany)[(size_t)m * Ndim + n] = (_Float16)x;
                } else {
                    ((float*)Outany)[(size_t)m * Ndim + n] = (float)(_Float16)x;
                }
            }
        }
    }
}

extern "C" void kernel_launch(void* const* d_in, const int* in_sizes, int n_in,
                              void* d_out, int out_size, void* d_ws, size_t ws_size,
                              hipStream_t stream) {
    const float* x  = (const float*)d_in[0];   // (B,L,H) fp32
    const float* W0 = (const float*)d_in[1];   // (N,H)   fp32
    const float* b0 = (const float*)d_in[2];   // (N)
    const float* W1 = (const float*)d_in[3];   // (H,N)   fp32
    const float* b1 = (const float*)d_in[4];   // (H)

    const int H = in_sizes[4];                 // 2048
    const int N = in_sizes[2];                 // 8192
    const int M = in_sizes[0] / H;             // 4096

    const size_t nX  = (size_t)M * H;          //  8.4M
    const size_t nW0 = (size_t)N * H;          // 16.8M
    const size_t nW1 = (size_t)H * N;          // 16.8M
    const size_t nD  = (size_t)M * N;          // 33.6M

    _Float16* D = (_Float16*)d_ws;             // [0, 64MB): fp16 intermediate
    const size_t needFast = (nD + nX + nW0 + nW1) * sizeof(_Float16);

    dim3 blk(NTHREADS);
    dim3 g0(N / BN, M / BM);                   // (64, 32)
    dim3 g1(H / BN, M / BM);                   // (16, 32)

    if (ws_size >= needFast) {
        // Fast path: pre-convert everything to fp16 once, then fully-async GEMMs.
        _Float16* Xh  = D + nD;
        _Float16* W0h = Xh + nX;
        _Float16* W1h = W0h + nW0;
        cvt_f32_f16<<<dim3((unsigned)(nX  / 4 / 256)), blk, 0, stream>>>(x,  Xh,  (int)(nX  / 4));
        cvt_f32_f16<<<dim3((unsigned)(nW0 / 4 / 256)), blk, 0, stream>>>(W0, W0h, (int)(nW0 / 4));
        cvt_f32_f16<<<dim3((unsigned)(nW1 / 4 / 256)), blk, 0, stream>>>(W1, W1h, (int)(nW1 / 4));
        mlp_gemm_nt<true, true, true ><<<g0, blk, 0, stream>>>(Xh, W0h, b0, D,     H, N);
        mlp_gemm_nt<true, true, false><<<g1, blk, 0, stream>>>(D,  W1h, b1, d_out, N, H);
    } else {
        // Fallback: convert fp32 tiles in-loop (needs only the D intermediate).
        mlp_gemm_nt<false, false, true ><<<g0, blk, 0, stream>>>(x, W0, b0, D,     H, N);
        mlp_gemm_nt<true,  false, false><<<g1, blk, 0, stream>>>(D, W1, b1, d_out, N, H);
    }
}
